// HungarianMatcher_80358838108633
// MI455X (gfx1250) — matrix operand order repaired and checked by reading, not compiled, not run on hardware
//
#include <hip/hip_runtime.h>
#include <hip/hip_bf16.h>

typedef __attribute__((ext_vector_type(16))) _Float16 v16h;
typedef __attribute__((ext_vector_type(8)))  _Float16 v8h;
typedef __attribute__((ext_vector_type(8)))  float    v8f;

#define BS    32
#define NQ    1024
#define NCLS  91
#define KPAD  96     // ncls padded to multiple of 32 for K-steps (pad cols are 0 in A -> exact)
#define NGT   256
#define LSA_INF 3.0e38f

// ---------------------------------------------------------------------------
// Phase 1: S[q, c] = pos(q,c) - neg(q,c) for batch-0 logits, f16, K-padded.
// ---------------------------------------------------------------------------
__global__ void build_s_kernel(const float* __restrict__ logits, _Float16* __restrict__ S) {
  int idx = blockIdx.x * blockDim.x + threadIdx.x;
  if (idx >= NQ * KPAD) return;
  int q = idx / KPAD, c = idx % KPAD;
  float s = 0.0f;
  if (c < NCLS) {
    float x = logits[q * NCLS + c];
    float p = 1.0f / (1.0f + __expf(-x));
    float om = 1.0f - p;
    float neg = 0.75f * p * p * (-__logf(om + 1e-8f));   // (1-ALPHA)*p^2*(-log(1-p))
    float pos = 0.25f * om * om * (-__logf(p + 1e-8f));  // ALPHA*(1-p)^2*(-log p)
    s = pos - neg;
  }
  S[idx] = (_Float16)s;
}

// ---------------------------------------------------------------------------
// Phase 2: fused cost build.  One wave per 16x16 (q,g) tile.
// Class term via v_wmma_f32_16x16x32_f16 against a register-built one-hot B.
// L1 + GIoU fused per-lane using the 16x16 f32 C/D layout (lane=N, vgpr=M).
// Stores TRANSPOSED cost CT[b, g, q] so the LAP solver reads contiguous rows.
// ---------------------------------------------------------------------------
__global__ void cost_kernel(const _Float16* __restrict__ S,
                            const float* __restrict__ pred_boxes,  // [bs,nq,4], batch 0 used
                            const int*   __restrict__ tgt_labels,  // [bs,ngt]
                            const float* __restrict__ tgt_boxes,   // [bs,ngt,4]
                            float* __restrict__ CT)                // [bs,ngt,nq]
{
  const int wave   = threadIdx.x >> 5;
  const int lane   = threadIdx.x & 31;
  const int tile   = blockIdx.x * (blockDim.x >> 5) + wave;   // 0 .. 32*64*16-1
  const int b      = tile >> 10;
  const int rem    = tile & 1023;
  const int qt     = rem >> 4;    // 0..63
  const int gt     = rem & 15;    // 0..15
  const int lane16 = lane & 15;
  const int hi     = lane >> 4;   // 0/1 half-wave

  const int q_a = qt * 16 + lane16;          // A row for this lane
  const int g   = gt * 16 + lane16;          // B/C/D column for this lane
  const int id  = tgt_labels[b * NGT + g];   // this lane's one-hot class

  v8f acc = {};
  #pragma unroll
  for (int kk = 0; kk < KPAD; kk += 32) {
    // A: f16 16x32 layout — lo lanes K=kk..kk+7 & kk+16..23; hi lanes +8
    const int ka = kk + hi * 8;
    v8h alo = *(const v8h*)(S + q_a * KPAD + ka);        // global_load_b128
    v8h ahi = *(const v8h*)(S + q_a * KPAD + ka + 16);   // global_load_b128
    v16h A;
    #pragma unroll
    for (int t = 0; t < 8; ++t) { A[t] = alo[t]; A[8 + t] = ahi[t]; }
    // B: f16 32x16 layout — element i of lane L is K = kk + 8*hi + i (i<8), kk + 16 + 8*hi + (i-8) (i>=8), N = L%16
    const int kb = kk + hi * 8;
    v16h Bm;
    #pragma unroll
    for (int t = 0; t < 16; ++t)
      Bm[t] = (id == kb + (t < 8 ? t : t + 8)) ? (_Float16)1.0f : (_Float16)0.0f;
    acc = __builtin_amdgcn_wmma_f32_16x16x32_f16(
        /*neg_a=*/false, A, /*neg_b=*/false, Bm,
        /*c_mod=*/(short)0, acc, /*reuse_a=*/false, /*reuse_b=*/false);
  }

  // Target box (fixed per lane): cxcywh -> xyxy
  const float* tb = tgt_boxes + (size_t)(b * NGT + g) * 4;
  const float tcx = tb[0], tcy = tb[1], tw = tb[2], th = tb[3];
  const float tx0 = tcx - 0.5f * tw, ty0 = tcy - 0.5f * th;
  const float tx1 = tcx + 0.5f * tw, ty1 = tcy + 0.5f * th;
  const float area2 = (tx1 - tx0) * (ty1 - ty0);

  #pragma unroll
  for (int r = 0; r < 8; ++r) {
    const int q = qt * 16 + r + hi * 8;          // D layout: M = r + 8*hi
    const float* pb = pred_boxes + (size_t)q * 4; // batch 0
    const float cx = pb[0], cy = pb[1], w = pb[2], h = pb[3];
    const float l1 = fabsf(cx - tcx) + fabsf(cy - tcy) + fabsf(w - tw) + fabsf(h - th);
    const float px0 = cx - 0.5f * w, py0 = cy - 0.5f * h;
    const float px1 = cx + 0.5f * w, py1 = cy + 0.5f * h;
    const float area1 = (px1 - px0) * (py1 - py0);
    const float ix0 = fmaxf(px0, tx0), iy0 = fmaxf(py0, ty0);
    const float ix1 = fminf(px1, tx1), iy1 = fminf(py1, ty1);
    const float inter = fmaxf(ix1 - ix0, 0.0f) * fmaxf(iy1 - iy0, 0.0f);
    const float uni = area1 + area2 - inter;
    const float iou = inter / uni;
    const float ex0 = fminf(px0, tx0), ey0 = fminf(py0, ty0);
    const float ex1 = fmaxf(px1, tx1), ey1 = fmaxf(py1, ty1);
    const float enc = fmaxf(ex1 - ex0, 0.0f) * fmaxf(ey1 - ey0, 0.0f);
    const float giou = iou - (enc - uni) / enc;
    const float c = 1.0f * acc[r] + 5.0f * l1 + 2.0f * (-giou);
    CT[((size_t)b * NGT + g) * NQ + q] = c;
  }
}

// ---------------------------------------------------------------------------
// Phase 3: Jonker-Volgenant LAP, one 1024-thread workgroup per image.
// Solves the 256x1024 (gt x pred) problem; all state in LDS (~31 KB).
// Thread j owns column j; argmin via LDS tree reduction.
// Finishes with in-block rank computation and float output writes.
// ---------------------------------------------------------------------------
__global__ void __launch_bounds__(1024)
lsa_kernel(const float* __restrict__ CT, float* __restrict__ out) {
  const int b = blockIdx.x;
  const int j = threadIdx.x;
  const float* cost = CT + (size_t)b * NGT * NQ;   // [256 rows][1024 cols]

  __shared__ float u_s[NGT];
  __shared__ float v_s[NQ];
  __shared__ int   row4col[NQ];
  __shared__ int   col4row[NGT];
  __shared__ int   remaining[NQ];
  __shared__ float spc[NQ];
  __shared__ int   path[NQ];
  __shared__ int   SR[NGT];
  __shared__ float redv[NQ];
  __shared__ int   redi[NQ];
  __shared__ int   sh_i, sh_sink, sh_done;
  __shared__ float sh_min;

  v_s[j] = 0.0f; row4col[j] = -1;
  if (j < NGT) { u_s[j] = 0.0f; col4row[j] = -1; }
  __syncthreads();

  for (int cur = 0; cur < NGT; ++cur) {
    remaining[j] = 1; spc[j] = LSA_INF; path[j] = -1;
    if (j < NGT) SR[j] = 0;
    if (j == 0) { sh_i = cur; sh_min = 0.0f; sh_done = 0; sh_sink = -1; }
    __syncthreads();

    while (true) {
      if (j == 0) SR[sh_i] = 1;
      __syncthreads();
      const int   i  = sh_i;
      const float mv = sh_min;
      const float rr = mv + cost[i * NQ + j] - u_s[i] - v_s[j];
      if (remaining[j] && rr < spc[j]) { spc[j] = rr; path[j] = i; }
      redv[j] = remaining[j] ? spc[j] : LSA_INF;
      redi[j] = j;
      __syncthreads();
      for (int s = NQ >> 1; s > 0; s >>= 1) {
        if (j < s && redv[j + s] < redv[j]) { redv[j] = redv[j + s]; redi[j] = redi[j + s]; }
        __syncthreads();
      }
      if (j == 0) {
        const int jm = redi[0];
        sh_min = redv[0];
        remaining[jm] = 0;
        const int r = row4col[jm];
        if (r < 0) { sh_sink = jm; sh_done = 1; } else sh_i = r;
      }
      __syncthreads();
      if (sh_done) break;
    }

    // Dual updates (scipy order: before augmentation)
    const float mv = sh_min;
    if (j < NGT && SR[j]) {
      if (j == cur) u_s[j] += mv;
      else          u_s[j] += mv - spc[col4row[j]];
    }
    if (!remaining[j]) v_s[j] -= mv - spc[j];   // SC = scanned columns
    __syncthreads();

    // Augment alternating path back to cur (serial, thread 0)
    if (j == 0) {
      int jj = sh_sink;
      while (true) {
        const int i = path[jj];
        row4col[jj] = i;
        const int jn = col4row[i];
        col4row[i] = jj;
        if (i == cur) break;
        jj = jn;
      }
    }
    __syncthreads();
  }

  // Rank by prediction index (distinct values -> bijective ranks 0..255)
  if (j < NGT) {
    const int c4r = col4row[j];
    int rank = 0;
    for (int g = 0; g < NGT; ++g) rank += (col4row[g] < c4r) ? 1 : 0;
    out[b * NGT + rank]            = (float)c4r;  // indices_i (pred index)
    out[BS * NGT + b * NGT + rank] = (float)j;    // indices_j (gt index)
  }
}

// ---------------------------------------------------------------------------
extern "C" void kernel_launch(void* const* d_in, const int* in_sizes, int n_in,
                              void* d_out, int out_size, void* d_ws, size_t ws_size,
                              hipStream_t stream) {
  const float* pred_logits = (const float*)d_in[0];  // [32,1024,91]
  const float* pred_boxes  = (const float*)d_in[1];  // [32,1024,4]
  const int*   tgt_labels  = (const int*)d_in[2];    // [32,256]
  const float* tgt_boxes   = (const float*)d_in[3];  // [32,256,4]
  float* out = (float*)d_out;                        // 2 * 32 * 256 floats

  // Workspace: CT (32 MB f32, transposed cost) then S (192 KB f16 focal diff)
  float*    CT = (float*)d_ws;
  _Float16* S  = (_Float16*)((char*)d_ws + (size_t)BS * NGT * NQ * sizeof(float));

  build_s_kernel<<<(NQ * KPAD + 255) / 256, 256, 0, stream>>>(pred_logits, S);

  // 32 batches * 64 q-tiles * 16 g-tiles = 32768 tiles, 8 waves per 256-thread block
  cost_kernel<<<(BS * 64 * 16) / 8, 256, 0, stream>>>(S, pred_boxes, tgt_labels, tgt_boxes, CT);

  lsa_kernel<<<BS, NQ, 0, stream>>>(CT, out);
}
